// SepMamba_40802189312119
// MI455X (gfx1250) — matrix operand order repaired and checked by reading, not compiled, not run on hardware
//
#include <hip/hip_runtime.h>
#include <stdint.h>

// ---------------------------------------------------------------------------
// SepMamba forward for gfx1250 (MI455X, wave32).
// Dense projections: V_WMMA_F32_16X16X32_BF16. Operands pre-padded (M->64,
// N->32, K->32 multiples) by the fp32->bf16 conversion kernels so the GEMM
// is branch-free; K loop is software-pipelined (register double buffering)
// so global_load_b128 of block k+1 overlaps the 8 wmma of block k.
// ---------------------------------------------------------------------------

typedef __bf16 bf16_t;
typedef __attribute__((ext_vector_type(16))) __bf16 v16bf;
typedef __attribute__((ext_vector_type(8)))  __bf16 v8bf;
typedef __attribute__((ext_vector_type(8)))  float  v8f;

static __device__ __forceinline__ bf16_t f2bf(float f) {
  unsigned u = __float_as_uint(f);
  unsigned r = u + 0x7FFFu + ((u >> 16) & 1u);           // round-to-nearest-even
  unsigned short h = (unsigned short)(r >> 16);
  return __builtin_bit_cast(bf16_t, h);
}

#define SHUF16(lo, hi) __builtin_shufflevector((lo), (hi), 0, 1, 2, 3, 4, 5, 6, 7, \
                                               8, 9, 10, 11, 12, 13, 14, 15)
#define WMMA_BF16(a, b, c) __builtin_amdgcn_wmma_f32_16x16x32_bf16(false, (a), false, (b), \
                                                                   (short)0, (c), false, false)

// ---------------------- padded fp32 -> bf16 conversion ---------------------
// dst is [rows_pad, cols_pad] bf16, zero-filled outside [rows, cols).
__global__ void k_cvt_pad(const float* __restrict__ s, long sstride, long rows, int cols,
                          bf16_t* __restrict__ d, int cols_pad, long total) {
  long i = (long)blockIdx.x * blockDim.x + threadIdx.x;
  if (i >= total) return;
  int c  = (int)(i % cols_pad);
  long r = i / cols_pad;
  float v = (r < rows && c < cols) ? s[r * sstride + c] : 0.f;
  d[i] = f2bf(v);
}

__global__ void k_zerof(float* p, int n) {
  int i = blockIdx.x * blockDim.x + threadIdx.x;
  if (i < n) p[i] = 0.f;
}

// ------------------------------ WMMA NT-GEMM -------------------------------
// C[Mp,Np] = A[Mp,Kp] * W[Np,Kp]^T (+ bias[Np]).  Pre-padded:
// Kp % 32 == 0, Np % 32 == 0, Mp % 64 == 0.  One wave per 64x32 C tile
// (4 row tiles x 2 col tiles, 8 accumulators). Double-buffered K loop.
__global__ void k_gemm_nt_wmma(const bf16_t* __restrict__ A, const bf16_t* __restrict__ W,
                               const float* __restrict__ bias, float* __restrict__ C,
                               int Kp, int Np) {
  const int  lane = threadIdx.x;          // blockDim == 32 (one wave)
  const int  col  = lane & 15;
  const bool hi   = lane >= 16;
  const int  n0   = blockIdx.x * 32;
  const int  m0   = blockIdx.y * 64;
  const int  oflo = hi ? 16 : 0, ofhi = hi ? 24 : 8;
  v8f acc[8] = {};

  const bf16_t* w0 = W + (long)(n0 + col) * Kp + (hi ? 16 : 0);
  const bf16_t* w1 = w0 + (long)16 * Kp;
  const bf16_t* arow[4];
#pragma unroll
  for (int rt = 0; rt < 4; ++rt)
    arow[rt] = A + (long)(m0 + rt * 16 + col) * Kp;

  // preload K block 0
  v16bf b0 = *(const v16bf*)(w0);
  v16bf b1 = *(const v16bf*)(w1);
  v8bf alo[4], ahi[4];
#pragma unroll
  for (int rt = 0; rt < 4; ++rt) {
    alo[rt] = *(const v8bf*)(arow[rt] + oflo);
    ahi[rt] = *(const v8bf*)(arow[rt] + ofhi);
  }

  int k0 = 0;
  for (; k0 + 32 < Kp; k0 += 32) {
    // prefetch next K block while computing on current
    v16bf nb0 = *(const v16bf*)(w0 + k0 + 32);
    v16bf nb1 = *(const v16bf*)(w1 + k0 + 32);
    v8bf nalo[4], nahi[4];
#pragma unroll
    for (int rt = 0; rt < 4; ++rt) {
      nalo[rt] = *(const v8bf*)(arow[rt] + k0 + 32 + oflo);
      nahi[rt] = *(const v8bf*)(arow[rt] + k0 + 32 + ofhi);
    }
#pragma unroll
    for (int rt = 0; rt < 4; ++rt) {
      v16bf afr = SHUF16(alo[rt], ahi[rt]);
      acc[rt]     = WMMA_BF16(afr, b0, acc[rt]);
      acc[4 + rt] = WMMA_BF16(afr, b1, acc[4 + rt]);
    }
    b0 = nb0; b1 = nb1;
#pragma unroll
    for (int rt = 0; rt < 4; ++rt) { alo[rt] = nalo[rt]; ahi[rt] = nahi[rt]; }
  }
  // final K block
#pragma unroll
  for (int rt = 0; rt < 4; ++rt) {
    v16bf afr = SHUF16(alo[rt], ahi[rt]);
    acc[rt]     = WMMA_BF16(afr, b0, acc[rt]);
    acc[4 + rt] = WMMA_BF16(afr, b1, acc[4 + rt]);
  }

  const float bv0 = (bias != nullptr) ? bias[n0 + col]      : 0.0f;
  const float bv1 = (bias != nullptr) ? bias[n0 + 16 + col] : 0.0f;
#pragma unroll
  for (int rt = 0; rt < 4; ++rt) {
    long mrow = (long)(m0 + rt * 16 + (hi ? 8 : 0));
#pragma unroll
    for (int r = 0; r < 8; ++r) {
      C[(mrow + r) * Np + n0 + col]      = acc[rt][r]     + bv0;
      C[(mrow + r) * Np + n0 + 16 + col] = acc[4 + rt][r] + bv1;
    }
  }
}

// ------------------------------- convolutions ------------------------------

__global__ void k_conv1d(const float* __restrict__ x, const float* __restrict__ w,
                         const float* __restrict__ b, float* __restrict__ y,
                         int B, int Cin, int Cout, int Tin, int Tout,
                         int ksz, int stride, int pad) {
  long idx = (long)blockIdx.x * blockDim.x + threadIdx.x;
  long total = (long)B * Cout * Tout;
  if (idx >= total) return;
  int t  = (int)(idx % Tout);
  long r = idx / Tout;
  int co = (int)(r % Cout);
  int bb = (int)(r / Cout);
  float acc = b[co];
  int tbase = t * stride - pad;
  for (int ci = 0; ci < Cin; ++ci) {
    const float* xp = x + ((long)bb * Cin + ci) * Tin;
    const float* wp = w + ((long)co * Cin + ci) * ksz;
    for (int j = 0; j < ksz; ++j) {
      int ts = tbase + j;
      if (ts >= 0 && ts < Tin) acc += xp[ts] * wp[j];
    }
  }
  y[idx] = acc;
}

// torch ConvTranspose1d, w layout [Cin, Cout, k]; only first Tout outputs computed
__global__ void k_convT1d(const float* __restrict__ x, const float* __restrict__ w,
                          const float* __restrict__ b, float* __restrict__ y,
                          int B, int Cin, int Cout, int Tin, int Tout,
                          int ksz, int stride, int pad) {
  long idx = (long)blockIdx.x * blockDim.x + threadIdx.x;
  long total = (long)B * Cout * Tout;
  if (idx >= total) return;
  int t  = (int)(idx % Tout);
  long r = idx / Tout;
  int co = (int)(r % Cout);
  int bb = (int)(r / Cout);
  float acc = b[co];
  for (int j = 0; j < ksz; ++j) {
    int tmp = t + pad - j;
    if (tmp < 0 || (tmp % stride) != 0) continue;
    int s = tmp / stride;
    if (s >= Tin) continue;
    for (int ci = 0; ci < Cin; ++ci)
      acc += x[((long)bb * Cin + ci) * Tin + s] * w[((long)ci * Cout + co) * ksz + j];
  }
  y[idx] = acc;
}

// ------------------------- GroupNorm(1,C) + PReLU --------------------------

__global__ void k_gn_stats(const float* __restrict__ x, float* __restrict__ stats, long CT) {
  int bb = blockIdx.y;
  long base = (long)bb * CT;
  float s = 0.f, q = 0.f;
  for (long i = (long)blockIdx.x * blockDim.x + threadIdx.x; i < CT;
       i += (long)gridDim.x * blockDim.x) {
    float v = x[base + i];
    s += v; q += v * v;
  }
  __shared__ float rs[256], rq[256];
  int tid = threadIdx.x;
  rs[tid] = s; rq[tid] = q;
  __syncthreads();
  for (int o = 128; o > 0; o >>= 1) {
    if (tid < o) { rs[tid] += rs[tid + o]; rq[tid] += rq[tid + o]; }
    __syncthreads();
  }
  if (tid == 0) {
    atomicAdd(&stats[2 * bb],     rs[0]);
    atomicAdd(&stats[2 * bb + 1], rq[0]);
  }
}

// chan_last==0: x is [B,C,T]; chan_last==1: x is [B,T,C]
__global__ void k_gn_prelu(const float* __restrict__ x, float* __restrict__ y,
                           const float* __restrict__ stats, const float* __restrict__ g,
                           const float* __restrict__ beta, const float* __restrict__ alpha,
                           int C, int T, long CT, int chan_last, long total) {
  long idx = (long)blockIdx.x * blockDim.x + threadIdx.x;
  if (idx >= total) return;
  int bb   = (int)(idx / CT);
  long rem = idx % CT;
  int c = chan_last ? (int)(rem % C) : (int)(rem / T);
  float mean = stats[2 * bb] / (float)CT;
  float var  = stats[2 * bb + 1] / (float)CT - mean * mean;
  float v = (x[idx] - mean) * rsqrtf(var + 1e-5f) * g[c] + beta[c];
  float a = alpha[0];
  y[idx] = v > 0.f ? v : a * v;
}

// ------------------------------- LayerNorm ---------------------------------

__global__ void k_ln(const float* __restrict__ x, float* __restrict__ y,
                     const float* __restrict__ g, const float* __restrict__ b,
                     long rows, int d) {
  long r = (long)blockIdx.x * blockDim.x + threadIdx.x;
  if (r >= rows) return;
  const float* xp = x + r * d;
  float* yp = y + r * d;
  float s = 0.f;
  for (int i = 0; i < d; ++i) s += xp[i];
  float m = s / (float)d;
  float v = 0.f;
  for (int i = 0; i < d; ++i) { float t = xp[i] - m; v += t * t; }
  float inv = rsqrtf(v / (float)d + 1e-5f);
  for (int i = 0; i < d; ++i) yp[i] = (xp[i] - m) * inv * g[i] + b[i];
}

// -------------------------- layout transforms ------------------------------

__global__ void k_tr_c2l(const float* __restrict__ x, float* __restrict__ y,
                         int B, int C, int T) {
  long idx = (long)blockIdx.x * blockDim.x + threadIdx.x;
  long total = (long)B * C * T;
  if (idx >= total) return;
  int c  = (int)(idx % C);
  long r = idx / C;
  int t  = (int)(r % T);
  int bb = (int)(r / T);
  y[idx] = x[((long)bb * C + c) * T + t];
}

__global__ void k_tr_l2c(const float* __restrict__ x, float* __restrict__ y,
                         int B, int C, int T) {
  long idx = (long)blockIdx.x * blockDim.x + threadIdx.x;
  long total = (long)B * C * T;
  if (idx >= total) return;
  int t  = (int)(idx % T);
  long r = idx / T;
  int c  = (int)(r % C);
  int bb = (int)(r / C);
  y[idx] = x[((long)bb * T + t) * C + c];
}

__global__ void k_copy_chan(const float* __restrict__ src, float* __restrict__ dst,
                            int B, int Cs, int T, int Cd, int coff) {
  long idx = (long)blockIdx.x * blockDim.x + threadIdx.x;
  long total = (long)B * Cs * T;
  if (idx >= total) return;
  int t  = (int)(idx % T);
  long r = idx / T;
  int c  = (int)(r % Cs);
  int bb = (int)(r / Cs);
  dst[((long)bb * Cd + coff + c) * T + t] = src[idx];
}

// ------------------------- mamba pieces (fp32) -----------------------------

// xz: [rows, 2*di] row-major (xm = cols [0,di)); causal depthwise conv k=4 + SiLU
__global__ void k_dwconv_silu(const float* __restrict__ xz, const float* __restrict__ w,
                              const float* __restrict__ bias, float* __restrict__ u,
                              int B, int T, int di) {
  long idx = (long)blockIdx.x * blockDim.x + threadIdx.x;
  long total = (long)B * T * di;
  if (idx >= total) return;
  int c  = (int)(idx % di);
  long r = idx / di;
  int t  = (int)(r % T);
  int bb = (int)(r / T);
  float acc = bias[c];
#pragma unroll
  for (int j = 0; j < 4; ++j) {
    int ts = t - 3 + j;
    if (ts >= 0) acc += xz[((long)bb * T + ts) * (2 * di) + c] * w[c * 4 + j];
  }
  u[idx] = acc / (1.f + __expf(-acc));
}

__global__ void k_softplus(float* __restrict__ y, long n) {
  long i = (long)blockIdx.x * blockDim.x + threadIdx.x;
  if (i >= n) return;
  float v = y[i];
  y[i] = (v > 20.f) ? v : log1pf(__expf(v));
}

// sequential SSM scan: one thread per (b, channel); B_/C_ broadcast via LDS
__global__ void k_ssm_scan(const float* __restrict__ delta, const float* __restrict__ u,
                           const float* __restrict__ proj, const float* __restrict__ A_log,
                           const float* __restrict__ D, float* __restrict__ y,
                           int T, int di, int pstride, int dtr) {
  int c  = threadIdx.x;          // blockDim == di
  int bb = blockIdx.y;
  __shared__ float sBC[32];
  float A[16], h[16];
#pragma unroll
  for (int n = 0; n < 16; ++n) { A[n] = -__expf(A_log[c * 16 + n]); h[n] = 0.f; }
  float Dc = D[c];
  for (int t = 0; t < T; ++t) {
    long rb = (long)bb * T + t;
    if (threadIdx.x < 32) sBC[threadIdx.x] = proj[rb * pstride + dtr + threadIdx.x];
    __syncthreads();
    float dl = delta[rb * di + c];
    float uu = u[rb * di + c];
    float acc = 0.f;
#pragma unroll
    for (int n = 0; n < 16; ++n) {
      float dA = __expf(dl * A[n]);
      h[n] = dA * h[n] + dl * sBC[n] * uu;
      acc += h[n] * sBC[16 + n];
    }
    y[rb * di + c] = acc + uu * Dc;
    __syncthreads();
  }
}

__global__ void k_gate(const float* __restrict__ y, const float* __restrict__ xz,
                       float* __restrict__ g, long rows, int di) {
  long idx = (long)blockIdx.x * blockDim.x + threadIdx.x;
  long total = rows * (long)di;
  if (idx >= total) return;
  int c  = (int)(idx % di);
  long r = idx / di;
  float z = xz[r * (2 * di) + di + c];
  g[idx] = y[idx] * z / (1.f + __expf(-z));
}

__global__ void k_flipT(const float* __restrict__ x, float* __restrict__ y,
                        int B, int T, int d) {
  long idx = (long)blockIdx.x * blockDim.x + threadIdx.x;
  long total = (long)B * T * d;
  if (idx >= total) return;
  int c  = (int)(idx % d);
  long r = idx / d;
  int t  = (int)(r % T);
  int bb = (int)(r / T);
  y[((long)bb * T + (T - 1 - t)) * d + c] = x[idx];
}

__global__ void k_residual(const float* __restrict__ x, const float* __restrict__ f,
                           const float* __restrict__ bw, float* __restrict__ out,
                           int B, int T, int d) {
  long idx = (long)blockIdx.x * blockDim.x + threadIdx.x;
  long total = (long)B * T * d;
  if (idx >= total) return;
  int c  = (int)(idx % d);
  long r = idx / d;
  int t  = (int)(r % T);
  int bb = (int)(r / T);
  out[idx] = x[idx] + f[idx] + bw[((long)bb * T + (T - 1 - t)) * d + c];
}

// ------------------------------- host side ---------------------------------

struct MambaP {
  const float *A_log, *D, *conv_b, *conv_w, *dt_bias, *dt_proj, *in_proj, *out_proj, *x_proj;
  int d, di, dtr;
};
struct BiP { MambaP bwd, fwd; const float *ln_b, *ln_g; };

extern "C" void kernel_launch(void* const* d_in, const int* in_sizes, int n_in,
                              void* d_out, int out_size, void* d_ws, size_t ws_size,
                              hipStream_t stream) {
  (void)in_sizes; (void)n_in; (void)out_size; (void)ws_size;

  const int B = 4;
  const int T0 = 16384, T1 = 8193, T2 = 4097, T3 = 2049;

  // ---- unpack inputs (JAX pytree order: dict keys sorted at every level) ----
  int ci = 0;
  auto nx = [&]() -> const float* { return (const float*)d_in[ci++]; };
  const float* mixture = nx();

  auto rd_mamba = [&](int d) {
    MambaP m;
    m.A_log = nx(); m.D = nx(); m.conv_b = nx(); m.conv_w = nx();
    m.dt_bias = nx(); m.dt_proj = nx(); m.in_proj = nx(); m.out_proj = nx(); m.x_proj = nx();
    m.d = d; m.di = 2 * d; m.dtr = (d + 15) / 16;
    return m;
  };
  auto rd_bi = [&](int d) {
    BiP b; b.bwd = rd_mamba(d); b.fwd = rd_mamba(d); b.ln_b = nx(); b.ln_g = nx();
    return b;
  };

  BiP bott[2];
  for (int i = 0; i < 2; ++i) bott[i] = rd_bi(256);

  struct DecP { const float *a, *beta, *fuse_b, *fuse_w, *g, *up_b, *up_w; BiP mam[2]; };
  DecP dec[2];
  const int dec_d[2] = {128, 64};
  for (int i = 0; i < 2; ++i) {
    dec[i].a = nx(); dec[i].beta = nx(); dec[i].fuse_b = nx(); dec[i].fuse_w = nx();
    dec[i].g = nx();
    dec[i].mam[0] = rd_bi(dec_d[i]); dec[i].mam[1] = rd_bi(dec_d[i]);
    dec[i].up_b = nx(); dec[i].up_w = nx();
  }

  struct EncP { const float *a, *beta, *conv_b, *conv_w, *g; BiP mam[2]; };
  EncP enc[2];
  const int enc_d[2] = {128, 256};
  for (int i = 0; i < 2; ++i) {
    enc[i].a = nx(); enc[i].beta = nx(); enc[i].conv_b = nx(); enc[i].conv_w = nx();
    enc[i].g = nx();
    enc[i].mam[0] = rd_bi(enc_d[i]); enc[i].mam[1] = rd_bi(enc_d[i]);
  }
  const float* in_a    = nx();
  const float* in_b    = nx();
  const float* in_beta = nx();
  const float* in_g    = nx();
  const float* in_w    = nx();
  const float* out_b   = nx();
  const float* out_w   = nx();

  // ---- workspace bump allocator ----
  char* base = (char*)d_ws;
  size_t off = 0;
  auto allocf = [&](size_t n) { float*  p = (float*) (base + off); off += ((n * 4 + 255) / 256) * 256; return p; };
  auto allocb = [&](size_t n) { bf16_t* p = (bf16_t*)(base + off); off += ((n * 2 + 255) / 256) * 256; return p; };

  float* stats = allocf(16);
  float* skip0 = allocf((size_t)B * 64  * T1);
  float* skip1 = allocf((size_t)B * 128 * T2);
  float* bufA  = allocf(6600000);
  float* bufB  = allocf(6600000);
  float* hb    = allocf(2200000);
  float* hf    = allocf(2200000);
  float* mf    = allocf(2200000);   // GEMM out, padded rows
  float* mb    = allocf(2200000);   // GEMM out, padded rows
  float* xz    = allocf(8600000);   // GEMM out, padded rows
  float* ub    = allocf(4300000);
  float* pj    = allocf(2200000);   // GEMM out, stride 64, padded rows
  float* dl    = allocf(4300000);   // GEMM out, padded rows
  float* yb    = allocf(4300000);
  float* gt    = allocf(4300000);
  bf16_t* hbf   = allocb(2200000);  // [Mp, d]
  bf16_t* ubf   = allocb(4300000);  // [Mp, di]
  bf16_t* dtbf  = allocb(1100000);  // [Mp, 32]
  bf16_t* gbf   = allocb(4300000);  // [Mp, di]
  bf16_t* wbf   = allocb(300000);   // [Np, Kp]
  bf16_t* bigbf = allocb(6600000);  // fuse activations [Mp, K]

  // ---- launch helpers ----
  auto ewg = [](long n) { return dim3((unsigned)((n + 255) / 256)); };
  auto cvtpad = [&](const float* s, long sstride, long rows, int cols,
                    bf16_t* d, long rows_pad, int cols_pad) {
    long total = rows_pad * (long)cols_pad;
    k_cvt_pad<<<ewg(total), 256, 0, stream>>>(s, sstride, rows, cols, d, cols_pad, total);
  };
  auto gemm = [&](const bf16_t* A, const bf16_t* W, const float* bias, float* C,
                  long Mp, int Np, int Kp) {
    dim3 g((unsigned)(Np / 32), (unsigned)(Mp / 64));
    k_gemm_nt_wmma<<<g, 32, 0, stream>>>(A, W, bias, C, Kp, Np);
  };
  auto gn = [&](const float* x, float* y, const float* g, const float* beta,
                const float* a, int C, int T, int chan_last) {
    k_zerof<<<1, 32, 0, stream>>>(stats, 2 * B);
    long CT = (long)C * T;
    k_gn_stats<<<dim3(64, B), 256, 0, stream>>>(x, stats, CT);
    long total = (long)B * CT;
    k_gn_prelu<<<ewg(total), 256, 0, stream>>>(x, y, stats, g, beta, a, C, T, CT,
                                               chan_last, total);
  };

  auto mamba = [&](const float* xin, float* out, int T, const MambaP& p) {
    const int d = p.d, di = p.di, dtr = p.dtr, pc = dtr + 32;
    const long BT = (long)B * T;
    const long Mp = (BT + 63) / 64 * 64;
    // xz = x @ in_proj^T                      [BT,d] x [2di,d]^T
    cvtpad(xin, d, BT, d, hbf, Mp, d);
    cvtpad(p.in_proj, d, 2 * di, d, wbf, 2 * di, d);
    gemm(hbf, wbf, nullptr, xz, Mp, 2 * di, d);
    // u = silu(depthwise_causal_conv(xm) + conv_b)
    k_dwconv_silu<<<ewg(BT * di), 256, 0, stream>>>(xz, p.conv_w, p.conv_b, ub, B, T, di);
    // proj = u @ x_proj^T                     N padded pc -> 64 (zeros), stride 64
    cvtpad(ub, di, BT, di, ubf, Mp, di);
    cvtpad(p.x_proj, di, pc, di, wbf, 64, di);
    gemm(ubf, wbf, nullptr, pj, Mp, 64, di);
    // delta = softplus(dt @ dt_proj^T + dt_bias)   K padded dtr -> 32 (zeros)
    cvtpad(pj, 64, BT, dtr, dtbf, Mp, 32);
    cvtpad(p.dt_proj, dtr, di, dtr, wbf, di, 32);
    gemm(dtbf, wbf, p.dt_bias, dl, Mp, di, 32);
    k_softplus<<<ewg(BT * di), 256, 0, stream>>>(dl, BT * di);
    // fused selective scan
    k_ssm_scan<<<dim3(1, B), dim3(di), 0, stream>>>(dl, ub, pj, p.A_log, p.D, yb,
                                                    T, di, 64, dtr);
    // gate + out projection
    k_gate<<<ewg(BT * di), 256, 0, stream>>>(yb, xz, gt, BT, di);
    cvtpad(gt, di, BT, di, gbf, Mp, di);
    cvtpad(p.out_proj, di, d, di, wbf, d, di);
    gemm(gbf, wbf, nullptr, out, Mp, d, di);
  };

  auto bimamba = [&](float* x, int T, const BiP& bp) {
    const int d = bp.fwd.d;
    const long BT = (long)B * T;
    k_ln<<<ewg(BT), 256, 0, stream>>>(x, hb, bp.ln_g, bp.ln_b, BT, d);
    mamba(hb, mf, T, bp.fwd);
    k_flipT<<<ewg(BT * d), 256, 0, stream>>>(hb, hf, B, T, d);
    mamba(hf, mb, T, bp.bwd);
    k_residual<<<ewg(BT * d), 256, 0, stream>>>(x, mf, mb, x, B, T, d);
  };

  // ------------------------------ forward -------------------------------
  // input conv + GN + PReLU                      [4,1,16384] -> [4,64,8193]
  k_conv1d<<<ewg((long)B * 64 * T1), 256, 0, stream>>>(mixture, in_w, in_b, bufA,
                                                       B, 1, 64, T0, T1, 16, 2, 8);
  gn(bufA, bufA, in_g, in_beta, in_a, 64, T1, 0);
  hipMemcpyAsync(skip0, bufA, (size_t)B * 64 * T1 * 4, hipMemcpyDeviceToDevice, stream);

  // encoder stage 0: 64 -> 128, T1 -> T2
  k_conv1d<<<ewg((long)B * 128 * T2), 256, 0, stream>>>(bufA, enc[0].conv_w, enc[0].conv_b,
                                                        bufB, B, 64, 128, T1, T2, 16, 2, 8);
  gn(bufB, bufB, enc[0].g, enc[0].beta, enc[0].a, 128, T2, 0);
  k_tr_c2l<<<ewg((long)B * 128 * T2), 256, 0, stream>>>(bufB, bufA, B, 128, T2);
  bimamba(bufA, T2, enc[0].mam[0]);
  bimamba(bufA, T2, enc[0].mam[1]);
  k_tr_l2c<<<ewg((long)B * 128 * T2), 256, 0, stream>>>(bufA, bufB, B, 128, T2);
  hipMemcpyAsync(skip1, bufB, (size_t)B * 128 * T2 * 4, hipMemcpyDeviceToDevice, stream);

  // encoder stage 1: 128 -> 256, T2 -> T3
  k_conv1d<<<ewg((long)B * 256 * T3), 256, 0, stream>>>(bufB, enc[1].conv_w, enc[1].conv_b,
                                                        bufA, B, 128, 256, T2, T3, 16, 2, 8);
  gn(bufA, bufA, enc[1].g, enc[1].beta, enc[1].a, 256, T3, 0);
  k_tr_c2l<<<ewg((long)B * 256 * T3), 256, 0, stream>>>(bufA, bufB, B, 256, T3);
  bimamba(bufB, T3, enc[1].mam[0]);
  bimamba(bufB, T3, enc[1].mam[1]);

  // bottleneck (channel-last)
  bimamba(bufB, T3, bott[0]);
  bimamba(bufB, T3, bott[1]);
  k_tr_l2c<<<ewg((long)B * 256 * T3), 256, 0, stream>>>(bufB, bufA, B, 256, T3);

  // decoder stage 0: upsample 256@T3 -> 256@T2, concat skip1, fuse -> 128
  k_convT1d<<<ewg((long)B * 256 * T2), 256, 0, stream>>>(bufA, dec[0].up_w, dec[0].up_b,
                                                         bufB, B, 256, 256, T3, T2, 16, 2, 8);
  k_copy_chan<<<ewg((long)B * 256 * T2), 256, 0, stream>>>(bufB, bufA, B, 256, T2, 384, 0);
  k_copy_chan<<<ewg((long)B * 128 * T2), 256, 0, stream>>>(skip1, bufA, B, 128, T2, 384, 256);
  k_tr_c2l<<<ewg((long)B * 384 * T2), 256, 0, stream>>>(bufA, bufB, B, 384, T2);
  {
    long BT2 = (long)B * T2, Mp2 = (BT2 + 63) / 64 * 64;
    cvtpad(bufB, 384, BT2, 384, bigbf, Mp2, 384);
    cvtpad(dec[0].fuse_w, 384, 128, 384, wbf, 128, 384);
    gemm(bigbf, wbf, dec[0].fuse_b, bufA, Mp2, 128, 384);   // channel-last out
  }
  gn(bufA, bufA, dec[0].g, dec[0].beta, dec[0].a, 128, T2, 1);
  bimamba(bufA, T2, dec[0].mam[0]);
  bimamba(bufA, T2, dec[0].mam[1]);
  k_tr_l2c<<<ewg((long)B * 128 * T2), 256, 0, stream>>>(bufA, bufB, B, 128, T2);

  // decoder stage 1: upsample 128@T2 -> 128@T1, concat skip0, fuse -> 64
  k_convT1d<<<ewg((long)B * 128 * T1), 256, 0, stream>>>(bufB, dec[1].up_w, dec[1].up_b,
                                                         bufA, B, 128, 128, T2, T1, 16, 2, 8);
  k_copy_chan<<<ewg((long)B * 128 * T1), 256, 0, stream>>>(bufA, bufB, B, 128, T1, 192, 0);
  k_copy_chan<<<ewg((long)B * 64 * T1), 256, 0, stream>>>(skip0, bufB, B, 64, T1, 192, 128);
  k_tr_c2l<<<ewg((long)B * 192 * T1), 256, 0, stream>>>(bufB, bufA, B, 192, T1);
  {
    long BT1 = (long)B * T1, Mp1 = (BT1 + 63) / 64 * 64;
    cvtpad(bufA, 192, BT1, 192, bigbf, Mp1, 192);
    cvtpad(dec[1].fuse_w, 192, 64, 192, wbf, 64, 192);
    gemm(bigbf, wbf, dec[1].fuse_b, bufB, Mp1, 64, 192);
  }
  gn(bufB, bufB, dec[1].g, dec[1].beta, dec[1].a, 64, T1, 1);
  bimamba(bufB, T1, dec[1].mam[0]);
  bimamba(bufB, T1, dec[1].mam[1]);
  k_tr_l2c<<<ewg((long)B * 64 * T1), 256, 0, stream>>>(bufB, bufA, B, 64, T1);

  // output transposed conv: [4,64,8193] -> [4,2,16384]
  k_convT1d<<<ewg((long)B * 2 * T0), 256, 0, stream>>>(bufA, out_w, out_b, (float*)d_out,
                                                       B, 64, 2, T1, T0, 16, 2, 8);
}
